// Com_CNN_RNN_18021682774631
// MI455X (gfx1250) — compile-verified, matrix operand order checked by reading, not run on hardware
//
#include <hip/hip_runtime.h>

typedef __attribute__((ext_vector_type(16))) __bf16 v16bf;
typedef __attribute__((ext_vector_type(8)))  float  v8f;
typedef __attribute__((ext_vector_type(2)))  float  v2f;

#define E  512
#define T  256
#define H3 1536
#define NT 96        // 1536 / 16 N-tiles

#if __has_builtin(__builtin_amdgcn_tensor_load_to_lds) && __has_builtin(__builtin_amdgcn_s_wait_tensorcnt)
#define HAVE_TDM 1
#else
#define HAVE_TDM 0
#endif

__device__ __forceinline__ unsigned short f2bf(float f) {
    unsigned int u = __builtin_bit_cast(unsigned int, f);
    u += 0x7FFFu + ((u >> 16) & 1u);            // round-to-nearest-even
    return (unsigned short)(u >> 16);
}
__device__ __forceinline__ float bf2f(unsigned short h) {
    unsigned int u = ((unsigned int)h) << 16;
    return __builtin_bit_cast(float, u);
}
__device__ __forceinline__ v2f bfpair(unsigned int w) {
    v2f r;
    r.x = bf2f((unsigned short)(w & 0xFFFFu));
    r.y = bf2f((unsigned short)(w >> 16));
    return r;
}
__device__ __forceinline__ float sigmoidf_(float x) { return 1.0f / (1.0f + __expf(-x)); }

#if HAVE_TDM
typedef unsigned int tdm_v4u __attribute__((ext_vector_type(4)));
typedef int          tdm_v8i __attribute__((ext_vector_type(8)));
typedef int          tdm_v4i __attribute__((ext_vector_type(4)));

// Async-load one 1536-float row (6 KB) from global into LDS via the Tensor Data Mover.
// D# built per CDNA5 ISA 8.3/8.4: group0 = {count=1 | lds_addr | global_addr | type=2},
// group1 = {data_size=4B, tensor_dim0=1536, tensor_dim1=1, tile_dim0=1536, tile_dim1=1,
//           tensor_dim0_stride=1536}.
__device__ __forceinline__ void tdm_load_row6k(const float* src, unsigned int lds_byte_off) {
    unsigned long long ga = (unsigned long long)(size_t)src;
    tdm_v4u g0;
    g0.x = 1u;                                                    // count=1, no gather
    g0.y = lds_byte_off;                                          // lds_addr
    g0.z = (unsigned int)ga;                                      // global_addr[31:0]
    g0.w = ((unsigned int)(ga >> 32) & 0x01FFFFFFu) | (2u << 30); // addr[56:32] | type=2
    tdm_v8i g1;
    g1[0] = 0x20000;              // workgroup_mask=0, data_size=2 (4 bytes)
    g1[1] = (int)(1536u << 16);   // tensor_dim0[15:0] in bits 63:48
    g1[2] = (int)(1u << 16);      // tensor_dim0 hi = 0 ; tensor_dim1 lo = 1
    g1[3] = (int)(1536u << 16);   // tensor_dim1 hi = 0 ; tile_dim0 = 1536
    g1[4] = 1;                    // tile_dim1 = 1, tile_dim2 = 0
    g1[5] = 1536;                 // tensor_dim0_stride lo32
    g1[6] = 0;
    g1[7] = 0;
    tdm_v4i gz = {0, 0, 0, 0};
#if defined(__clang_major__) && __clang_major__ >= 23
    tdm_v8i gz8 = {0, 0, 0, 0, 0, 0, 0, 0};
    __builtin_amdgcn_tensor_load_to_lds(g0, g1, gz, gz, gz8, 0);
#else
    __builtin_amdgcn_tensor_load_to_lds(g0, g1, gz, gz, 0);
#endif
}
#endif

// ---------------- elementwise prep kernels ----------------

__global__ void k_f32_to_bf16(const float* __restrict__ in, unsigned short* __restrict__ out, int n) {
    int i = blockIdx.x * blockDim.x + threadIdx.x;
    if (i < n) out[i] = f2bf(in[i]);
}

__global__ void k_zero_bf(unsigned short* __restrict__ p, int n) {
    int i = blockIdx.x * blockDim.x + threadIdx.x;
    if (i < n) p[i] = 0;
}

__global__ void k_embed(const int* __restrict__ sA, const int* __restrict__ sB,
                        const float* __restrict__ emb, unsigned short* __restrict__ X) {
    int i = blockIdx.x * blockDim.x + threadIdx.x;
    if (i >= 2 * T * E) return;
    int k = i & (E - 1);
    int t = (i >> 9) & (T - 1);
    int s = i >> 17;
    int tok = (s == 0) ? sA[t] : sB[t];
    X[i] = f2bf(emb[(size_t)tok * E + k]);
}

__global__ void k_pack_ep2(const float* __restrict__ fin, unsigned short* __restrict__ X) {
    int i = blockIdx.x * blockDim.x + threadIdx.x;
    if (i >= 2 * 16 * E) return;
    int k = i & (E - 1);
    int r = (i >> 9) & 15;
    int s = i >> 13;
    X[i] = (r < 2) ? f2bf(fin[(s * 2 + r) * E + k]) : (unsigned short)0;
}

// ---------------- WMMA GEMM: C[M,1536] = A[M,512] * W[1536,512]^T + bias ----------------
// 2x2 register tiling: each wave owns a 32x32 output block -> 4 fragment loads feed
// 4 v_wmma_f32_16x16x32_bf16 per K-step (2x the MACC/byte of a single-tile wave).

__global__ void k_gemm_wmma(const unsigned short* __restrict__ A,   // [Mpad][512] bf16
                            const unsigned short* __restrict__ W,   // [1536][512] bf16
                            const float* __restrict__ bias,         // [1536]
                            float* __restrict__ C,                  // [M][1536] f32
                            int Mtotal) {
    int wave = (int)(((unsigned)blockIdx.x * blockDim.x + threadIdx.x) >> 5);
    int lane = threadIdx.x & 31;
    int Mtiles = (Mtotal + 15) >> 4;
    int Mg = (Mtiles + 1) >> 1;
    const int Ng = NT / 2;                 // 48 N-groups of 32 columns
    if (wave >= Mg * Ng) return;
    int m0 = (wave / Ng) * 32, n0 = (wave % Ng) * 32;
    int half = lane >> 4, lr = lane & 15;

    const unsigned short* a0p = A + (size_t)(m0 + lr) * E + half * 16;
    const unsigned short* a1p = a0p + (size_t)16 * E;
    const unsigned short* b0p = W + (size_t)(n0 + lr) * E + half * 16;
    const unsigned short* b1p = b0p + (size_t)16 * E;

    v8f c00 = {}, c01 = {}, c10 = {}, c11 = {};
#pragma unroll
    for (int kk = 0; kk < E; kk += 32) {
        v16bf a0 = *reinterpret_cast<const v16bf*>(a0p + kk);
        v16bf a1 = *reinterpret_cast<const v16bf*>(a1p + kk);
        v16bf b0 = *reinterpret_cast<const v16bf*>(b0p + kk);
        v16bf b1 = *reinterpret_cast<const v16bf*>(b1p + kk);
        c00 = __builtin_amdgcn_wmma_f32_16x16x32_bf16(false, a0, false, b0, (short)0, c00, false, false);
        c01 = __builtin_amdgcn_wmma_f32_16x16x32_bf16(false, a0, false, b1, (short)0, c01, false, false);
        c10 = __builtin_amdgcn_wmma_f32_16x16x32_bf16(false, a1, false, b0, (short)0, c10, false, false);
        c11 = __builtin_amdgcn_wmma_f32_16x16x32_bf16(false, a1, false, b1, (short)0, c11, false, false);
    }
    int nA = n0 + lr, nB = nA + 16;
    float bnA = bias[nA], bnB = bias[nB];
#pragma unroll
    for (int j = 0; j < 8; ++j) {          // VGPR j: M=j (lanes 0-15) / M=8+j (lanes 16-31)
        int mA = m0 + j + 8 * half;
        int mB = mA + 16;
        if (mA < Mtotal) {
            C[(size_t)mA * H3 + nA] = c00[j] + bnA;
            C[(size_t)mA * H3 + nB] = c01[j] + bnB;
        }
        if (mB < Mtotal) {
            C[(size_t)mB * H3 + nA] = c10[j] + bnA;
            C[(size_t)mB * H3 + nB] = c11[j] + bnB;
        }
    }
}

// ---------------- sequential GRU scan: 1 block per sentence, h in LDS ----------------
// TDM double-buffers the next step's 6 KB gate row into LDS while the matvec runs;
// dots are float2-vectorized to invite v_pk_fma_f32 / ds_load_b64.

__global__ void __launch_bounds__(512)
k_gru_scan(const float* __restrict__ gi, int L, int Lpad,
           const unsigned short* __restrict__ Whh,   // [1536][512] bf16
           const float* __restrict__ bhh,            // [1536]
           unsigned short* __restrict__ seq, int LpadOut,
           float* __restrict__ hfin, int sentStride) {
    __shared__ float h[E];
#if HAVE_TDM
    __shared__ float gbuf[2][H3];
#endif
    int s = blockIdx.x;
    int j = threadIdx.x;          // 512 threads: thread j owns rows j, 512+j, 1024+j
    h[j] = 0.0f;
#if HAVE_TDM
    if (threadIdx.x == 0) {
        tdm_load_row6k(gi + (size_t)(s * Lpad) * H3, (unsigned int)(size_t)(void*)&gbuf[0][0]);
        __builtin_amdgcn_s_wait_tensorcnt(0);
    }
#endif
    __syncthreads();

    const uint4* wr = reinterpret_cast<const uint4*>(Whh + (size_t)j * E);
    const uint4* wz = reinterpret_cast<const uint4*>(Whh + (size_t)(E + j) * E);
    const uint4* wn = reinterpret_cast<const uint4*>(Whh + (size_t)(2 * E + j) * E);
    float br = bhh[j], bz = bhh[E + j], bn = bhh[2 * E + j];
    const v2f* h2 = reinterpret_cast<const v2f*>(h);

    for (int t = 0; t < L; ++t) {
#if HAVE_TDM
        if (threadIdx.x == 0 && t + 1 < L)
            tdm_load_row6k(gi + (size_t)(s * Lpad + t + 1) * H3,
                           (unsigned int)(size_t)(void*)&gbuf[(t + 1) & 1][0]);
        const float* g = gbuf[t & 1];
#else
        const float* g = gi + (size_t)(s * Lpad + t) * H3;
        if (t + 1 < L)
            __builtin_prefetch(gi + (size_t)(s * Lpad + t + 1) * H3 + j, 0, 1);
#endif
        v2f arv = {0.f, 0.f}, azv = {0.f, 0.f}, anv = {0.f, 0.f};
        for (int c = 0; c < E / 8; ++c) {
            v2f hp0 = h2[c * 4 + 0], hp1 = h2[c * 4 + 1];
            v2f hp2 = h2[c * 4 + 2], hp3 = h2[c * 4 + 3];
            uint4 a = wr[c];
            arv = __builtin_elementwise_fma(bfpair(a.x), hp0, arv);
            arv = __builtin_elementwise_fma(bfpair(a.y), hp1, arv);
            arv = __builtin_elementwise_fma(bfpair(a.z), hp2, arv);
            arv = __builtin_elementwise_fma(bfpair(a.w), hp3, arv);
            uint4 b = wz[c];
            azv = __builtin_elementwise_fma(bfpair(b.x), hp0, azv);
            azv = __builtin_elementwise_fma(bfpair(b.y), hp1, azv);
            azv = __builtin_elementwise_fma(bfpair(b.z), hp2, azv);
            azv = __builtin_elementwise_fma(bfpair(b.w), hp3, azv);
            uint4 d = wn[c];
            anv = __builtin_elementwise_fma(bfpair(d.x), hp0, anv);
            anv = __builtin_elementwise_fma(bfpair(d.y), hp1, anv);
            anv = __builtin_elementwise_fma(bfpair(d.z), hp2, anv);
            anv = __builtin_elementwise_fma(bfpair(d.w), hp3, anv);
        }
        float ar = arv.x + arv.y, az = azv.x + azv.y, an = anv.x + anv.y;
        float r  = sigmoidf_(g[j]         + ar + br);
        float z  = sigmoidf_(g[E + j]     + az + bz);
        float nn = tanhf   (g[2 * E + j] + r * (an + bn));
        float hn = (1.0f - z) * nn + z * h[j];
        __syncthreads();                        // all reads of old h (and gbuf) done
        h[j] = hn;
        seq[(size_t)(s * LpadOut + t) * E + j] = f2bf(hn);
#if HAVE_TDM
        if (threadIdx.x == 0 && t + 1 < L)
            __builtin_amdgcn_s_wait_tensorcnt(0);
#endif
        __syncthreads();
    }
    hfin[s * sentStride + j] = h[j];
}

// ---------------- tiny tail: conv1d + maxpool + GRU2 + bilinear head ----------------

__global__ void __launch_bounds__(512)
k_head(const float* __restrict__ hf,        // finals ep2: [2 sent][2 ch][512]
       const float* __restrict__ conv_w, const float* __restrict__ conv_b,
       const float* __restrict__ Wih2, const float* __restrict__ bih2,
       const float* __restrict__ Whh2, const float* __restrict__ bhh2,
       const float* __restrict__ WA, const float* __restrict__ WB,
       const float* __restrict__ b_bi, const float* __restrict__ W_lin,
       const float* __restrict__ b_lin, float* __restrict__ out) {
    __shared__ float conv[2][256];
    __shared__ float seq[2][128];
    __shared__ float gi2[2][H3];
    __shared__ float h2[E];
    __shared__ float hAB[2][E];
    __shared__ float ht[256];
    int tid = threadIdx.x;

    for (int s = 0; s < 2; ++s) {
        {   // conv1d: in=out=2 ch, kernel 512, stride 2, pad 255
            int c = tid >> 8, p = tid & 255;
            int start = p * 2 - 255;
            float acc = conv_b[c];
            for (int ic = 0; ic < 2; ++ic) {
                const float* x = hf + (size_t)(s * 2 + ic) * E;
                const float* w = conv_w + (size_t)(c * 2 + ic) * E;
                int k0 = start < 0 ? -start : 0;
                int k1 = start > 0 ? E - start : E;
                for (int k = k0; k < k1; ++k) acc = fmaf(x[start + k], w[k], acc);
            }
            conv[c][p] = acc;
        }
        __syncthreads();
        if (tid < 256) {   // maxpool kernel 512, stride 2, pad 255
            int c = tid >> 7, q = tid & 127;
            int start = q * 2 - 255;
            int p0 = start < 0 ? 0 : start;
            int p1 = (start + 511 < 255) ? start + 511 : 255;
            float m = -3.4e38f;
            for (int p = p0; p <= p1; ++p) m = fmaxf(m, conv[c][p]);
            seq[c][q] = m;
        }
        __syncthreads();
        for (int o = tid; o < 2 * H3; o += 512) {   // gi2 = seq @ Wih2^T + bih2
            int l = o / H3, jj = o % H3;
            float acc = bih2[jj];
            const float* w = Wih2 + (size_t)jj * 128;
            for (int k = 0; k < 128; ++k) acc = fmaf(w[k], seq[l][k], acc);
            gi2[l][jj] = acc;
        }
        h2[tid] = 0.0f;
        __syncthreads();
        for (int t = 0; t < 2; ++t) {   // GRU2, H=512, 2 steps
            float ar = 0.f, az = 0.f, an = 0.f;
            const float* wr = Whh2 + (size_t)tid * E;
            const float* wz = Whh2 + (size_t)(E + tid) * E;
            const float* wn = Whh2 + (size_t)(2 * E + tid) * E;
            for (int k = 0; k < E; ++k) {
                float hk = h2[k];
                ar = fmaf(wr[k], hk, ar);
                az = fmaf(wz[k], hk, az);
                an = fmaf(wn[k], hk, an);
            }
            float r  = sigmoidf_(gi2[t][tid]         + ar + bhh2[tid]);
            float z  = sigmoidf_(gi2[t][E + tid]     + az + bhh2[E + tid]);
            float nn = tanhf   (gi2[t][2 * E + tid] + r * (an + bhh2[2 * E + tid]));
            float hn = (1.0f - z) * nn + z * h2[tid];
            __syncthreads();
            h2[tid] = hn;
            __syncthreads();
        }
        hAB[s][tid] = h2[tid];
        __syncthreads();
    }
    if (tid < 256) {   // Double_Linear + tanh
        float acc = b_bi[tid];
        for (int k = 0; k < E; ++k) {
            float a = hAB[0][k], b = hAB[1][k];
            acc = fmaf(a * b,        WA[(size_t)k * 256 + tid], acc);
            acc = fmaf(fabsf(a - b), WB[(size_t)k * 256 + tid], acc);
        }
        ht[tid] = tanhf(acc);
    }
    __syncthreads();
    if (tid == 0) {
        float acc = b_lin[0];
        for (int u = 0; u < 256; ++u) acc = fmaf(ht[u], W_lin[u], acc);
        out[0] = sigmoidf_(acc);
    }
}

// ---------------- host launch ----------------

extern "C" void kernel_launch(void* const* d_in, const int* in_sizes, int n_in,
                              void* d_out, int out_size, void* d_ws, size_t ws_size,
                              hipStream_t stream) {
    (void)in_sizes; (void)n_in; (void)out_size; (void)ws_size;
    const int*   sentA  = (const int*)d_in[0];
    const int*   sentB  = (const int*)d_in[1];
    const float* emb    = (const float*)d_in[2];
    const float* Wih1   = (const float*)d_in[3];
    const float* Whh1   = (const float*)d_in[4];
    const float* bih1   = (const float*)d_in[5];
    const float* bhh1   = (const float*)d_in[6];
    const float* conv_w = (const float*)d_in[7];
    const float* conv_b = (const float*)d_in[8];
    const float* Wih2   = (const float*)d_in[9];
    const float* Whh2   = (const float*)d_in[10];
    const float* bih2   = (const float*)d_in[11];
    const float* bhh2   = (const float*)d_in[12];
    const float* WA     = (const float*)d_in[13];
    const float* WB     = (const float*)d_in[14];
    const float* b_bi   = (const float*)d_in[15];
    const float* W_lin  = (const float*)d_in[16];
    const float* b_lin  = (const float*)d_in[17];
    float* out = (float*)d_out;

    char* ws = (char*)d_ws;
    size_t off = 0;
    auto alloc = [&](size_t bytes) -> void* {
        off = (off + 255) & ~(size_t)255;
        void* p = ws + off;
        off += bytes;
        return p;
    };
    unsigned short* Xemb   = (unsigned short*)alloc((size_t)2 * T * E * 2);
    unsigned short* Wih1bf = (unsigned short*)alloc((size_t)2 * H3 * E * 2);
    unsigned short* Whh1bf = (unsigned short*)alloc((size_t)2 * H3 * E * 2);
    float*          gi     = (float*)alloc((size_t)2 * T * H3 * 4);
    unsigned short* S1     = (unsigned short*)alloc((size_t)2 * T * E * 2);
    unsigned short* S2     = (unsigned short*)alloc((size_t)2 * T * E * 2);
    float*          fin1   = (float*)alloc((size_t)2 * 2 * E * 4);
    unsigned short* Xa     = (unsigned short*)alloc((size_t)2 * 16 * E * 2);
    unsigned short* Xb     = (unsigned short*)alloc((size_t)2 * 16 * E * 2);
    float*          fin2   = (float*)alloc((size_t)2 * 2 * E * 4);

    const int nW = 2 * H3 * E;
    k_f32_to_bf16<<<(nW + 255) / 256, 256, 0, stream>>>(Wih1, Wih1bf, nW);
    k_f32_to_bf16<<<(nW + 255) / 256, 256, 0, stream>>>(Whh1, Whh1bf, nW);
    k_embed<<<(2 * T * E + 255) / 256, 256, 0, stream>>>(sentA, sentB, emb, Xemb);

    auto gemm = [&](const unsigned short* A, int layer, int Mtotal) {
        int Mtiles = (Mtotal + 15) / 16;
        int Mg = (Mtiles + 1) / 2;
        int waves  = Mg * (NT / 2);
        int blocks = (waves + 7) / 8;   // 256-thread block = 8 wave32
        k_gemm_wmma<<<blocks, 256, 0, stream>>>(A, Wih1bf + (size_t)layer * H3 * E,
                                                bih1 + layer * H3, gi, Mtotal);
    };

    // ---- epoch 1 (L = 256) ----
    gemm(Xemb, 0, 2 * T);
    k_gru_scan<<<2, 512, 0, stream>>>(gi, T, T, Whh1bf,                  bhh1,      S1, T, fin1,     2 * E);
    gemm(S1, 1, 2 * T);
    k_gru_scan<<<2, 512, 0, stream>>>(gi, T, T, Whh1bf + (size_t)H3 * E, bhh1 + H3, S2, T, fin1 + E, 2 * E);

    // ---- epoch 2 (L = 2, rows padded to 16 per sentence for WMMA tiles) ----
    k_pack_ep2<<<(2 * 16 * E + 255) / 256, 256, 0, stream>>>(fin1, Xa);
    k_zero_bf <<<(2 * 16 * E + 255) / 256, 256, 0, stream>>>(Xb, 2 * 16 * E);
    gemm(Xa, 0, 32);
    k_gru_scan<<<2, 512, 0, stream>>>(gi, 2, 16, Whh1bf,                  bhh1,      Xb, 16, fin2,     2 * E);
    gemm(Xb, 1, 32);
    k_gru_scan<<<2, 512, 0, stream>>>(gi, 2, 16, Whh1bf + (size_t)H3 * E, bhh1 + H3, S2, T, fin2 + E, 2 * E);

    // ---- conv + pool + GRU2 + head ----
    k_head<<<1, 512, 0, stream>>>(fin2, conv_w, conv_b, Wih2, bih2, Whh2, bhh2,
                                  WA, WB, b_bi, W_lin, b_lin, out);
}
